// MemoryBank_10703058502209
// MI455X (gfx1250) — matrix operand order repaired and checked by reading, not compiled, not run on hardware
//
#include <hip/hip_runtime.h>
#include <stdint.h>

typedef __attribute__((ext_vector_type(16))) _Float16 v16h;
typedef __attribute__((ext_vector_type(8)))  _Float16 v8h;
typedef __attribute__((ext_vector_type(8)))  float    v8f;

#define N_TRACKS 65536
#define L_MEM    4
#define DMODEL   256
#define HFFN     1024
#define NHEAD    8
#define DHEAD    32

// f16 weight layout in workspace (element offsets)
#define OFF_WIN   0            // in_proj  [768][256]  (Wq rows 0..255, Wk 256..511, Wv 512..767)
#define OFF_WOUT  196608       // out_proj [256][256]
#define OFF_WFC1  262144       // fc1      [1024][256]
#define OFF_WFC2  524288       // fc2      [256][1024]
#define OFF_WSAVE 786432       // save_proj[256][256]
#define TOT_WH    851968

// LDS strides (in elements)
#define SPh 264                // f16 stride for 256-wide tiles (16B-aligned rows, bank-skewed)
#define SHh 1032               // f16 stride for 1024-wide hidden tile
#define SF  264                // f32 stride

// LDS region byte offsets (with reuse across pipeline stages)
#define SM_A   0               // [64][SPh] f16 mem_bank tile  -> reused as hidden [16][SHh]
#define SM_K   33792           // [64][SPh] f16 K projections
#define SM_V   67584           // [64][SPh] f16 V projections
#define SM_D   101376          // [16][SPh] f16 E tile -> ctx -> eLN f16 -> new_emb f16
#define SM_E   109824          // [16][SF]  f32 q -> attn_out -> ffn_out -> save_embed
#define SM_AT  126720          // [16][8][4] f32 attn weights
#define SM_SV  128768          // int[16] saved flags
#define SM_SP  128832          // int[16] new save_period
#define SM_LN  128896          // [16][SF] f32 e-after-LN1 (residual for LN2)
#define SM_BYTES 145792

__global__ void prep_weights(const float* __restrict__ in_proj,
                             const float* __restrict__ out_proj,
                             const float* __restrict__ fc1,
                             const float* __restrict__ fc2,
                             const float* __restrict__ save_proj,
                             _Float16* __restrict__ wsh) {
  int i = blockIdx.x * blockDim.x + threadIdx.x;
  if (i >= TOT_WH) return;
  float v;
  if      (i < OFF_WOUT)  v = in_proj[i];
  else if (i < OFF_WFC1)  v = out_proj[i - OFF_WOUT];
  else if (i < OFF_WFC2)  v = fc1[i - OFF_WFC1];
  else if (i < OFF_WSAVE) v = fc2[i - OFF_WFC2];
  else                    v = save_proj[i - OFF_WSAVE];
  wsh[i] = (_Float16)v;
}

// A fragment (16x32 f16, row-major source): lane holds row (lane&15),
// K halves [g*8, g*8+7] and [16+g*8, 16+g*8+7]  (g = lane>>4)  -> two b128 reads.
__device__ __forceinline__ v16h load_a16(const _Float16* base, int strideH, int lane, int k0) {
  int r = lane & 15, g = lane >> 4;
  const _Float16* p = base + r * strideH + k0 + g * 8;
  v8h lo = *(const v8h*)p;
  v8h hi = *(const v8h*)(p + 16);
  return __builtin_shufflevector(lo, hi, 0,1,2,3,4,5,6,7,8,9,10,11,12,13,14,15);
}

// B fragment (32x16 f16) from weight matrix W[out][in] row-major:
// lane holds output col n = n0+(lane&15), K halves k0+[g*8, g*8+7] and k0+[16+g*8, 16+g*8+7] -> two b128 reads.
__device__ __forceinline__ v16h load_b16(const _Float16* w, int ldw, int n0, int k0, int lane) {
  const _Float16* p = w + (size_t)(n0 + (lane & 15)) * ldw + k0 + (lane >> 4) * 8;
  v8h lo = *(const v8h*)p;
  v8h hi = *(const v8h*)(p + 16);
  return __builtin_shufflevector(lo, hi, 0,1,2,3,4,5,6,7,8,9,10,11,12,13,14,15);
}

// One 16x16 f32 output tile: A from LDS f16, B from global f16 weights, K-loop of WMMAs.
__device__ __forceinline__ v8f gemm_tile(const _Float16* aLds, int lda,
                                         const _Float16* wB, int ldb,
                                         int n0, int K, int lane) {
  v8f c = {};
#pragma unroll 4
  for (int k0 = 0; k0 < K; k0 += 32) {
    v16h a = load_a16(aLds, lda, lane, k0);
    v16h b = load_b16(wB, ldb, n0, k0, lane);
    c = __builtin_amdgcn_wmma_f32_16x16x32_f16(false, a, false, b, (short)0, c, false, false);
  }
  return c;
}

__global__ __launch_bounds__(256)
void membank_kernel(const float* __restrict__ emb, const float* __restrict__ bank,
                    const float* __restrict__ scores, const unsigned char* __restrict__ mask,
                    const int* __restrict__ sper,
                    const float* __restrict__ b_in, const float* __restrict__ b_out,
                    const float* __restrict__ b_fc1, const float* __restrict__ b_fc2,
                    const float* __restrict__ b_save,
                    const float* __restrict__ ln1g, const float* __restrict__ ln1b,
                    const float* __restrict__ ln2g, const float* __restrict__ ln2b,
                    const _Float16* __restrict__ wsh,
                    float* __restrict__ outEmb, float* __restrict__ outBank,
                    float* __restrict__ outMask, float* __restrict__ outSp) {
  extern __shared__ unsigned char smem[];
  _Float16* shA    = (_Float16*)(smem + SM_A);
  _Float16* shK    = (_Float16*)(smem + SM_K);
  _Float16* shV    = (_Float16*)(smem + SM_V);
  _Float16* shD    = (_Float16*)(smem + SM_D);
  float*    shE    = (float*)(smem + SM_E);
  float*    shAttn = (float*)(smem + SM_AT);
  int*      shSaved= (int*)(smem + SM_SV);
  int*      shNSp  = (int*)(smem + SM_SP);
  float*    shLN   = (float*)(smem + SM_LN);

  const int t = threadIdx.x;
  const int lane = t & 31;
  const int wave = t >> 5;
  const int track0 = blockIdx.x * 16;

  // ---- Stage 0: stage E (16x256) and mem_bank (64x256) tiles into LDS as f16 ----
#pragma unroll
  for (int i = 0; i < 16; ++i) {
    int idx = i * 256 + t;
    int r = idx >> 8, c = idx & 255;
    shD[r * SPh + c] = (_Float16)emb[(size_t)(track0 + r) * DMODEL + c];
  }
#pragma unroll
  for (int i = 0; i < 64; ++i) {
    int idx = i * 256 + t;
    int r = idx >> 8, c = idx & 255;
    shA[r * SPh + c] = (_Float16)bank[((size_t)track0 * L_MEM + r) * DMODEL + c];
  }
  if (t < 16) {
    int sp = sper[track0 + t];
    int sv = (sp == 0) && (scores[track0 + t] > 0.4f);
    shSaved[t] = sv;
    shNSp[t] = sv ? 3 : (sp > 0 ? sp - 1 : sp);
  }
  __syncthreads();

  // ---- Stage 1: Q = E @ Wq^T + bq  -> shE (f32) ----
  for (int ti = 0; ti < 2; ++ti) {
    int n0 = (wave * 2 + ti) * 16;
    v8f c = gemm_tile(shD, SPh, wsh + OFF_WIN, DMODEL, n0, DMODEL, lane);
    int n = n0 + (lane & 15), g = lane >> 4;
    float bv = b_in[n];
#pragma unroll
    for (int i = 0; i < 8; ++i) shE[(i + 8 * g) * SF + n] = c[i] + bv;
  }
  __syncthreads();

  // ---- Stage 2: K,V = M @ {Wk,Wv}^T + b  -> shK/shV (f16).  128 tiles over 8 waves ----
  for (int j = 0; j < 16; ++j) {
    int T = wave * 16 + j;
    int isV = T >= 64;
    int T2 = T & 63;
    int r = T2 >> 4, n0 = (T2 & 15) * 16;
    const _Float16* wb = wsh + OFF_WIN + (size_t)(isV ? 512 : 256) * DMODEL;
    const float* bb = b_in + (isV ? 512 : 256);
    _Float16* dst = isV ? shV : shK;
    v8f c = gemm_tile(shA + r * 16 * SPh, SPh, wb, DMODEL, n0, DMODEL, lane);
    int n = n0 + (lane & 15), g = lane >> 4;
    float bv = bb[n];
#pragma unroll
    for (int i = 0; i < 8; ++i)
      dst[(r * 16 + i + 8 * g) * SPh + n] = (_Float16)(c[i] + bv);
  }
  __syncthreads();

  // ---- Stage 3: masked softmax over L=4 (one (track,head) per thread) ----
  if (t < 128) {
    int m = t >> 3, h = t & 7, d0 = h * DHEAD;
    int valid = mask[(size_t)(track0 + m) * L_MEM + (L_MEM - 1)] == 0;
    float lg[4];
    float mx = -3.0e38f;
#pragma unroll
    for (int l = 0; l < 4; ++l) {
      float acc = 0.f;
      const float* qp = shE + m * SF + d0;
      const _Float16* kp = shK + (m * 4 + l) * SPh + d0;
#pragma unroll
      for (int d = 0; d < DHEAD; ++d) acc += qp[d] * (float)kp[d];
      acc *= 0.17677669529663687f;  // 1/sqrt(32)
      int pad = mask[(size_t)(track0 + m) * L_MEM + l] != 0;
      if (pad && valid) acc = -3.0e38f;
      lg[l] = acc;
      mx = fmaxf(mx, acc);
    }
    float se = 0.f;
#pragma unroll
    for (int l = 0; l < 4; ++l) { lg[l] = __expf(lg[l] - mx); se += lg[l]; }
    float inv = 1.f / se;
#pragma unroll
    for (int l = 0; l < 4; ++l) shAttn[(m * 8 + h) * 4 + l] = lg[l] * inv;
  }
  __syncthreads();

  // ---- Stage 4: ctx = attn @ V  -> shD (f16, reuses E tile) ----
#pragma unroll
  for (int i = 0; i < 16; ++i) {
    int idx = i * 256 + t;
    int m = idx >> 8, d = idx & 255, h = d >> 5;
    const float* aw = shAttn + (m * 8 + h) * 4;
    float acc = 0.f;
#pragma unroll
    for (int l = 0; l < 4; ++l) acc += aw[l] * (float)shV[(m * 4 + l) * SPh + d];
    shD[m * SPh + d] = (_Float16)acc;
  }
  __syncthreads();

  // ---- Stage 5: attn_out = ctx @ Wo^T + bo -> shE (f32) ----
  for (int ti = 0; ti < 2; ++ti) {
    int n0 = (wave * 2 + ti) * 16;
    v8f c = gemm_tile(shD, SPh, wsh + OFF_WOUT, DMODEL, n0, DMODEL, lane);
    int n = n0 + (lane & 15), g = lane >> 4;
    float bv = b_out[n];
#pragma unroll
    for (int i = 0; i < 8; ++i) shE[(i + 8 * g) * SF + n] = c[i] + bv;
  }
  __syncthreads();

  // ---- Stage 6: LN1 over (E + attn_out): shfl_xor reduce within 16-lane half-wave ----
  {
    int row = t >> 4, sub = t & 15;
    float xv[16], s = 0.f, ss = 0.f;
#pragma unroll
    for (int j = 0; j < 16; ++j) {
      int c = sub + 16 * j;
      float x = emb[(size_t)(track0 + row) * DMODEL + c] + shE[row * SF + c];
      xv[j] = x; s += x; ss += x * x;
    }
#pragma unroll
    for (int m = 1; m <= 8; m <<= 1) { s += __shfl_xor(s, m, 32); ss += __shfl_xor(ss, m, 32); }
    float mu = s * (1.f / 256.f);
    float var = ss * (1.f / 256.f) - mu * mu;
    float rstd = rsqrtf(var + 1e-5f);
#pragma unroll
    for (int j = 0; j < 16; ++j) {
      int c = sub + 16 * j;
      float y = (xv[j] - mu) * rstd * ln1g[c] + ln1b[c];
      shLN[row * SF + c] = y;          // f32 residual for LN2
      shD[row * SPh + c] = (_Float16)y; // A operand for FFN1
    }
  }
  __syncthreads();

  // ---- Stage 7: hidden = relu(e @ fc1^T + b1) -> shA (f16 [16][SHh], reuses mem tile) ----
  for (int j = 0; j < 8; ++j) {
    int n0 = (wave * 8 + j) * 16;
    v8f c = gemm_tile(shD, SPh, wsh + OFF_WFC1, DMODEL, n0, DMODEL, lane);
    int n = n0 + (lane & 15), g = lane >> 4;
    float bv = b_fc1[n];
#pragma unroll
    for (int i = 0; i < 8; ++i) {
      float y = c[i] + bv;
      shA[(i + 8 * g) * SHh + n] = (_Float16)(y > 0.f ? y : 0.f);
    }
  }
  __syncthreads();

  // ---- Stage 8: ffn = hidden @ fc2^T + b2 -> shE (f32), K=1024 ----
  for (int ti = 0; ti < 2; ++ti) {
    int n0 = (wave * 2 + ti) * 16;
    v8f c = gemm_tile(shA, SHh, wsh + OFF_WFC2, HFFN, n0, HFFN, lane);
    int n = n0 + (lane & 15), g = lane >> 4;
    float bv = b_fc2[n];
#pragma unroll
    for (int i = 0; i < 8; ++i) shE[(i + 8 * g) * SF + n] = c[i] + bv;
  }
  __syncthreads();

  // ---- Stage 9: LN2, validity select, emit new_emb; stash f16 copy for save_proj ----
  {
    int row = t >> 4, sub = t & 15;
    int valid = mask[(size_t)(track0 + row) * L_MEM + (L_MEM - 1)] == 0;
    float xv[16], s = 0.f, ss = 0.f;
#pragma unroll
    for (int j = 0; j < 16; ++j) {
      int c = sub + 16 * j;
      float x = shLN[row * SF + c] + shE[row * SF + c];
      xv[j] = x; s += x; ss += x * x;
    }
#pragma unroll
    for (int m = 1; m <= 8; m <<= 1) { s += __shfl_xor(s, m, 32); ss += __shfl_xor(ss, m, 32); }
    float mu = s * (1.f / 256.f);
    float var = ss * (1.f / 256.f) - mu * mu;
    float rstd = rsqrtf(var + 1e-5f);
#pragma unroll
    for (int j = 0; j < 16; ++j) {
      int c = sub + 16 * j;
      float y = (xv[j] - mu) * rstd * ln2g[c] + ln2b[c];
      if (!valid) y = emb[(size_t)(track0 + row) * DMODEL + c];
      outEmb[(size_t)(track0 + row) * DMODEL + c] = y;
      shD[row * SPh + c] = (_Float16)y;
    }
  }
  __syncthreads();

  // ---- Stage 10: save_embed = new_emb @ Wsave^T + b -> shE (f32) ----
  for (int ti = 0; ti < 2; ++ti) {
    int n0 = (wave * 2 + ti) * 16;
    v8f c = gemm_tile(shD, SPh, wsh + OFF_WSAVE, DMODEL, n0, DMODEL, lane);
    int n = n0 + (lane & 15), g = lane >> 4;
    float bv = b_save[n];
#pragma unroll
    for (int i = 0; i < 8; ++i) shE[(i + 8 * g) * SF + n] = c[i] + bv;
  }
  __syncthreads();

  // ---- Stage 11: bank shift / mask / save_period update ----
#pragma unroll
  for (int i = 0; i < 64; ++i) {
    int idx = i * 256 + t;
    int m = idx >> 10, rem = idx & 1023, l = rem >> 8, d = rem & 255;
    size_t gRow = (size_t)(track0 + m) * L_MEM;
    int sv = shSaved[m];
    float v;
    if (sv) v = (l < 3) ? bank[(gRow + l + 1) * DMODEL + d] : shE[m * SF + d];
    else    v = bank[(gRow + l) * DMODEL + d];
    outBank[(gRow + l) * DMODEL + d] = v;
  }
  if (t < 64) {
    int m = t >> 2, l = t & 3;
    size_t gi = (size_t)(track0 + m) * L_MEM + l;
    int sv = shSaved[m];
    float nm;
    if (sv) nm = (l < 3) ? (mask[gi + 1] ? 1.f : 0.f) : 0.f;
    else    nm = mask[gi] ? 1.f : 0.f;
    outMask[gi] = nm;
  }
  if (t < 16) outSp[track0 + t] = (float)shNSp[t];
}

extern "C" void kernel_launch(void* const* d_in, const int* in_sizes, int n_in,
                              void* d_out, int out_size, void* d_ws, size_t ws_size,
                              hipStream_t stream) {
  (void)in_sizes; (void)n_in; (void)out_size; (void)ws_size;
  const float*         emb    = (const float*)d_in[0];
  const float*         bank   = (const float*)d_in[1];
  const float*         scores = (const float*)d_in[2];
  const unsigned char* mask   = (const unsigned char*)d_in[3];
  const int*           sper   = (const int*)d_in[4];
  const float* save_w = (const float*)d_in[5];
  const float* save_b = (const float*)d_in[6];
  const float* in_w   = (const float*)d_in[7];
  const float* in_b   = (const float*)d_in[8];
  const float* out_w  = (const float*)d_in[9];
  const float* out_b  = (const float*)d_in[10];
  const float* fc1_w  = (const float*)d_in[11];
  const float* fc1_b  = (const float*)d_in[12];
  const float* fc2_w  = (const float*)d_in[13];
  const float* fc2_b  = (const float*)d_in[14];
  const float* ln1g   = (const float*)d_in[15];
  const float* ln1b   = (const float*)d_in[16];
  const float* ln2g   = (const float*)d_in[17];
  const float* ln2b   = (const float*)d_in[18];

  float* o      = (float*)d_out;
  float* oEmb   = o;
  float* oBank  = oEmb + (size_t)N_TRACKS * DMODEL;
  float* oMask  = oBank + (size_t)N_TRACKS * L_MEM * DMODEL;
  float* oSp    = oMask + (size_t)N_TRACKS * L_MEM;

  _Float16* wsh = (_Float16*)d_ws;

  prep_weights<<<(TOT_WH + 255) / 256, 256, 0, stream>>>(in_w, out_w, fc1_w, fc2_w, save_w, wsh);

  membank_kernel<<<N_TRACKS / 16, 256, SM_BYTES, stream>>>(
      emb, bank, scores, mask, sper,
      in_b, out_b, fc1_b, fc2_b, save_b,
      ln1g, ln1b, ln2g, ln2b, wsh,
      oEmb, oBank, oMask, oSp);
}
